// Attention_17557826306356
// MI455X (gfx1250) — matrix operand order repaired and checked by reading, not compiled, not run on hardware
//
#include <hip/hip_runtime.h>
#include <hip/hip_bf16.h>
#include <math.h>

// ---------------- types for WMMA ----------------
typedef __bf16 bf16_t;
typedef bf16_t v16bf __attribute__((ext_vector_type(16)));
typedef float  v8f   __attribute__((ext_vector_type(8)));

#define B_    32
#define N_    1009
#define C_    768
#define H_    12
#define D_    64
#define M_    (B_*N_)        // 32288 rows
#define QKV_N 2304
#define VT_PITCH 1024        // padded key pitch for V^T (keys 1009..1023 zeroed)
#define BPITCH 40            // LDS B-tile pitch in shorts (80B: 16B aligned, conflict-free)

__device__ __forceinline__ unsigned short f2bf(float f){
    unsigned int x = __float_as_uint(f);
    x += 0x7FFFu + ((x >> 16) & 1u);      // round-to-nearest-even
    return (unsigned short)(x >> 16);
}
__device__ __forceinline__ unsigned int pack2bf(float lo, float hi){
    return (unsigned int)f2bf(lo) | ((unsigned int)f2bf(hi) << 16);
}

union AB { v16bf v; unsigned int u[8]; };

__device__ __forceinline__ v8f wmma_bf16(const AB& a, const AB& b, v8f c){
    return __builtin_amdgcn_wmma_f32_16x16x32_bf16(false, a.v, false, b.v,
                                                   (short)0, c, false, false);
}

// async 16B global->LDS copy (per-lane addresses), tracked by ASYNCcnt
__device__ __forceinline__ void async_copy16(unsigned lds_off, unsigned long long gaddr){
    asm volatile("global_load_async_to_lds_b128 %0, %1, off"
                 :: "v"(lds_off), "v"(gaddr) : "memory");
}

// ---------------- x -> bf16 ----------------
__global__ void k_conv_x(const float* __restrict__ x, unsigned short* __restrict__ xb){
    long long i = (long long)blockIdx.x * blockDim.x + threadIdx.x;
    const long long total = (long long)M_ * C_ / 4;
    if (i >= total) return;
    float4 f = ((const float4*)x)[i];
    ((uint2*)xb)[i] = make_uint2(pack2bf(f.x, f.y), pack2bf(f.z, f.w));
}

// ---------------- weight transpose + bf16 convert ----------------
__global__ void k_conv_w(const float* __restrict__ W, unsigned short* __restrict__ Wt,
                         int K, int Ncols){
    long long i = (long long)blockIdx.x * blockDim.x + threadIdx.x;
    if (i >= (long long)K * Ncols) return;
    int k = (int)(i / Ncols), n = (int)(i % Ncols);
    Wt[(long long)n * K + k] = f2bf(W[i]);
}

// zero the padded tail columns of V^T (keys 1009..1023)
__global__ void k_vpad(unsigned short* __restrict__ Vt){
    int i = blockIdx.x * blockDim.x + threadIdx.x;
    const int pad = VT_PITCH - N_;
    const int rows = B_ * H_ * D_;
    if (i >= rows * pad) return;
    int r = i / pad, c = i % pad;
    Vt[(long long)r * VT_PITCH + N_ + c] = 0;
}

// ---------------- QKV projection GEMM (bf16 A, async-LDS double-buffered B) ----------
__global__ __launch_bounds__(256) void k_qkv_gemm(const unsigned short* __restrict__ A,
        const unsigned short* __restrict__ Wt, const float* __restrict__ bias,
        unsigned short* __restrict__ Q, unsigned short* __restrict__ K,
        unsigned short* __restrict__ Vt){
    __shared__ __align__(16) unsigned short Bs[2 * 128 * BPITCH];
    const int lane = threadIdx.x & 31;
    const int wave = threadIdx.x >> 5;
    const int half = lane >> 4;
    const int ln   = lane & 15;
    const int m0   = blockIdx.x * 128 + wave * 16;
    const int n0   = blockIdx.y * 128;
    const int mrow = min(m0 + ln, M_ - 1);
    const unsigned lds0 = (unsigned)(size_t)(void*)Bs;

    auto fill = [&](int buf, int kk){
#pragma unroll
        for (int j = 0; j < 2; j++){
            int cid = (int)threadIdx.x + j * 256;      // 512 x 16B chunks = 128 rows x 64B
            int row = cid >> 2, c = cid & 3;
            unsigned loff = lds0 + (unsigned)(buf * (128*BPITCH) + row * BPITCH + c * 8) * 2u;
            unsigned long long g = (unsigned long long)(size_t)
                    (Wt + (size_t)(n0 + row) * C_ + kk + c * 8);
            async_copy16(loff, g);
        }
    };

    v8f acc[8];
    for (int t = 0; t < 8; t++) for (int v = 0; v < 8; v++) acc[t][v] = 0.f;

    fill(0, 0);
    const int NSTEP = C_ / 32;
    for (int step = 0; step < NSTEP; step++){
        const int cur = step & 1;
        __syncthreads();                                  // all reads of buf cur^1 done
        if (step + 1 < NSTEP){
            fill(cur ^ 1, (step + 1) * 32);
            asm volatile("s_wait_asynccnt 0x2" ::: "memory");   // cur's fills retired
        } else {
            asm volatile("s_wait_asynccnt 0x0" ::: "memory");
        }
        __syncthreads();                                  // cur visible to all waves
        AB a;
        {
            const unsigned short* ar = A + (size_t)mrow * C_ + step*32 + half*16;
            *(uint4*)&a.u[0] = *(const uint4*)ar;         // K = half*16 + 0..7
            *(uint4*)&a.u[4] = *(const uint4*)(ar + 8);   // K = half*16 + 8..15
        }
        const unsigned short* bs = Bs + cur * (128*BPITCH);
#pragma unroll
        for (int t = 0; t < 8; t++){
            AB bf;
            const unsigned short* row = bs + (t*16 + ln) * BPITCH + half*16;
            *(uint4*)&bf.u[0] = *(const uint4*)row;       // K = half*16 + 0..7
            *(uint4*)&bf.u[4] = *(const uint4*)(row + 8); // K = half*16 + 8..15
            acc[t] = wmma_bf16(a, bf, acc[t]);
        }
    }
    // epilogue: bias, bf16, scatter into Q / K / V^T
#pragma unroll
    for (int t = 0; t < 8; t++){
        int ccol = n0 + t*16 + ln;
        float bv = bias[ccol];
        int which = ccol / C_;
        int cc = ccol - which * C_;
        int h = cc >> 6, d = cc & 63;
#pragma unroll
        for (int v = 0; v < 8; v++){
            int crow = m0 + v + half * 8;
            if (crow >= M_) continue;
            int bidx = crow / N_, tok = crow - bidx * N_;
            int bh = bidx * H_ + h;
            unsigned short val = f2bf(acc[t][v] + bv);
            if (which == 0)      Q[((long long)bh * N_ + tok) * D_ + d] = val;
            else if (which == 1) K[((long long)bh * N_ + tok) * D_ + d] = val;
            else                 Vt[((long long)bh * D_ + d) * VT_PITCH + tok] = val;
        }
    }
}

// ---------------- prompt row: softmax over 1 key == copy V row 0 ----------------
__global__ void k_prompt(const unsigned short* __restrict__ Vt, unsigned short* __restrict__ Y){
    int i = blockIdx.x * blockDim.x + threadIdx.x;
    if (i >= B_ * C_) return;
    int b = i / C_, c = i % C_;
    int h = c >> 6, d = c & 63;
    Y[(long long)b * N_ * C_ + c] = Vt[(((long long)(b * H_ + h)) * D_ + d) * VT_PITCH];
}

// ---------------- flash attention: 1 wave per 16-query tile ----------------
__global__ __launch_bounds__(128) void k_attn(const unsigned short* __restrict__ Q,
        const unsigned short* __restrict__ K, const unsigned short* __restrict__ Vt,
        unsigned short* __restrict__ Y){
    __shared__ __align__(16) unsigned short lds[4 * 16 * BPITCH];
    const int lane = threadIdx.x & 31;
    const int wave = threadIdx.x >> 5;
    const int half = lane >> 4;
    const int ln   = lane & 15;

    int tile = blockIdx.x * 4 + wave;
    int bh = tile / 63;
    int tq = tile % 63;
    int qbase, kstart, kend, nkb;
    if (tq < 27){ qbase = 1 + tq * 16;          kstart = 1; kend = 433; nkb = 14; }
    else        { qbase = 433 + (tq - 27) * 16; kstart = 0; kend = N_;  nkb = 32; }

    const unsigned short* Qh = Q  + (long long)bh * N_ * D_;
    const unsigned short* Kh = K  + (long long)bh * N_ * D_;
    const unsigned short* Vh = Vt + (long long)bh * D_ * VT_PITCH;

    AB qa[2];
    {
        const unsigned short* qr = Qh + (long long)(qbase + ln) * D_;
#pragma unroll
        for (int c = 0; c < 2; c++){
            const unsigned short* p = qr + c*32 + half*16;
            *(uint4*)&qa[c].u[0] = *(const uint4*)p;
            *(uint4*)&qa[c].u[4] = *(const uint4*)(p + 8);
        }
    }

    float mi[8], li[8];
    v8f acc[4];
    for (int v = 0; v < 8; v++){ mi[v] = -1e30f; li[v] = 0.f; }
    for (int t = 0; t < 4; t++) for (int v = 0; v < 8; v++) acc[t][v] = 0.f;

    unsigned short* lp = lds + wave * 16 * BPITCH;
    const float scale = 0.125f;

    for (int ib = 0; ib < nkb; ib++){
        int kb = ib * 32;                              // always 32-aligned (alignment!)
        v8f s0, s1;
        for (int v = 0; v < 8; v++){ s0[v] = 0.f; s1[v] = 0.f; }
        int kc0 = min(kb + ln,      N_ - 1);
        int kc1 = min(kb + 16 + ln, N_ - 1);
#pragma unroll
        for (int c = 0; c < 2; c++){
            AB kf;
            const unsigned short* kr = Kh + (long long)kc0 * D_ + c*32 + half*16;
            *(uint4*)&kf.u[0] = *(const uint4*)kr;
            *(uint4*)&kf.u[4] = *(const uint4*)(kr + 8);
            s0 = wmma_bf16(qa[c], kf, s0);
        }
#pragma unroll
        for (int c = 0; c < 2; c++){
            AB kf;
            const unsigned short* kr = Kh + (long long)kc1 * D_ + c*32 + half*16;
            *(uint4*)&kf.u[0] = *(const uint4*)kr;
            *(uint4*)&kf.u[4] = *(const uint4*)(kr + 8);
            s1 = wmma_bf16(qa[c], kf, s1);
        }
        int col0 = kb + ln, col1 = kb + 16 + ln;
        bool msk0 = (col0 < kstart) || (col0 >= kend);
        bool msk1 = (col1 < kstart) || (col1 >= kend);
        float p0[8], p1[8], alpha[8];
#pragma unroll
        for (int v = 0; v < 8; v++){
            float a0 = msk0 ? -1e30f : s0[v] * scale;
            float a1 = msk1 ? -1e30f : s1[v] * scale;
            float rm = fmaxf(a0, a1);
            rm = fmaxf(rm, __shfl_xor(rm, 1, 32));
            rm = fmaxf(rm, __shfl_xor(rm, 2, 32));
            rm = fmaxf(rm, __shfl_xor(rm, 4, 32));
            rm = fmaxf(rm, __shfl_xor(rm, 8, 32));
            float mnew = fmaxf(mi[v], rm);
            alpha[v] = __expf(mi[v] - mnew);
            p0[v] = __expf(a0 - mnew);
            p1[v] = __expf(a1 - mnew);
            float rs = p0[v] + p1[v];
            rs += __shfl_xor(rs, 1, 32);
            rs += __shfl_xor(rs, 2, 32);
            rs += __shfl_xor(rs, 4, 32);
            rs += __shfl_xor(rs, 8, 32);
            li[v] = li[v] * alpha[v] + rs;
            mi[v] = mnew;
        }
#pragma unroll
        for (int t = 0; t < 4; t++)
#pragma unroll
            for (int v = 0; v < 8; v++) acc[t][v] *= alpha[v];
        // P (C layout) -> LDS row-major bf16 -> reload as A fragment
#pragma unroll
        for (int v = 0; v < 8; v++){
            int r = v + half * 8;
            lp[r*BPITCH + ln]      = f2bf(p0[v]);
            lp[r*BPITCH + 16 + ln] = f2bf(p1[v]);
        }
        asm volatile("s_wait_dscnt 0" ::: "memory");
        AB pa;
        {
            const unsigned short* pr = lp + ln*BPITCH + half*16;
            *(uint4*)&pa.u[0] = *(const uint4*)pr;
            *(uint4*)&pa.u[4] = *(const uint4*)(pr + 8);
        }
#pragma unroll
        for (int t = 0; t < 4; t++){
            AB vf;
            int d = t*16 + ln;
            const unsigned short* vr = Vh + (long long)d * VT_PITCH + kb + half*16;
            *(uint4*)&vf.u[0] = *(const uint4*)vr;
            *(uint4*)&vf.u[4] = *(const uint4*)(vr + 8);
            acc[t] = wmma_bf16(pa, vf, acc[t]);
        }
    }
    int b = bh / H_, h = bh - b * H_;
#pragma unroll
    for (int t = 0; t < 4; t++)
#pragma unroll
        for (int v = 0; v < 8; v++){
            int tok = qbase + v + half * 8;
            int d = t*16 + ln;
            Y[((long long)(b * N_ + tok)) * C_ + h * D_ + d] = f2bf(acc[t][v] / li[v]);
        }
}

// ---------------- output projection GEMM (same async-LDS scheme) ----------------
__global__ __launch_bounds__(256) void k_proj_gemm(const unsigned short* __restrict__ Y,
        const unsigned short* __restrict__ Wt, const float* __restrict__ bias,
        float* __restrict__ out){
    __shared__ __align__(16) unsigned short Bs[2 * 128 * BPITCH];
    const int lane = threadIdx.x & 31;
    const int wave = threadIdx.x >> 5;
    const int half = lane >> 4;
    const int ln   = lane & 15;
    const int m0   = blockIdx.x * 128 + wave * 16;
    const int n0   = blockIdx.y * 128;
    const int mrow = min(m0 + ln, M_ - 1);
    const unsigned lds0 = (unsigned)(size_t)(void*)Bs;

    auto fill = [&](int buf, int kk){
#pragma unroll
        for (int j = 0; j < 2; j++){
            int cid = (int)threadIdx.x + j * 256;
            int row = cid >> 2, c = cid & 3;
            unsigned loff = lds0 + (unsigned)(buf * (128*BPITCH) + row * BPITCH + c * 8) * 2u;
            unsigned long long g = (unsigned long long)(size_t)
                    (Wt + (size_t)(n0 + row) * C_ + kk + c * 8);
            async_copy16(loff, g);
        }
    };

    v8f acc[8];
    for (int t = 0; t < 8; t++) for (int v = 0; v < 8; v++) acc[t][v] = 0.f;

    fill(0, 0);
    const int NSTEP = C_ / 32;
    for (int step = 0; step < NSTEP; step++){
        const int cur = step & 1;
        __syncthreads();
        if (step + 1 < NSTEP){
            fill(cur ^ 1, (step + 1) * 32);
            asm volatile("s_wait_asynccnt 0x2" ::: "memory");
        } else {
            asm volatile("s_wait_asynccnt 0x0" ::: "memory");
        }
        __syncthreads();
        AB a;
        {
            const unsigned short* ar = Y + (size_t)mrow * C_ + step*32 + half*16;
            *(uint4*)&a.u[0] = *(const uint4*)ar;
            *(uint4*)&a.u[4] = *(const uint4*)(ar + 8);
        }
        const unsigned short* bs = Bs + cur * (128*BPITCH);
#pragma unroll
        for (int t = 0; t < 8; t++){
            AB bf;
            const unsigned short* row = bs + (t*16 + ln) * BPITCH + half*16;
            *(uint4*)&bf.u[0] = *(const uint4*)row;
            *(uint4*)&bf.u[4] = *(const uint4*)(row + 8);
            acc[t] = wmma_bf16(a, bf, acc[t]);
        }
    }
#pragma unroll
    for (int t = 0; t < 8; t++){
        int ccol = n0 + t*16 + ln;
        float bv = bias[ccol];
#pragma unroll
        for (int v = 0; v < 8; v++){
            int crow = m0 + v + half * 8;
            if (crow < M_) out[(long long)crow * C_ + ccol] = acc[t][v] + bv;
        }
    }
}

// ---------------- launch ----------------
extern "C" void kernel_launch(void* const* d_in, const int* in_sizes, int n_in,
                              void* d_out, int out_size, void* d_ws, size_t ws_size,
                              hipStream_t stream){
    const float* x      = (const float*)d_in[0];
    const float* W_qkv  = (const float*)d_in[1];
    const float* b_qkv  = (const float*)d_in[2];
    const float* W_proj = (const float*)d_in[3];
    const float* b_proj = (const float*)d_in[4];
    (void)in_sizes; (void)n_in; (void)out_size; (void)ws_size;

    char* ws = (char*)d_ws;
    size_t offQ  = 0;
    size_t offK  = offQ  + (size_t)B_*H_*N_*D_*2;        // Q bf16
    size_t offV  = offK  + (size_t)B_*H_*N_*D_*2;        // K bf16
    size_t offY  = offV  + (size_t)B_*H_*D_*VT_PITCH*2;  // V^T bf16 (padded)
    size_t offXb = offY  + (size_t)M_*C_*2;              // Y bf16
    size_t offWq = offXb + (size_t)M_*C_*2;              // x bf16
    size_t offWp = offWq + (size_t)QKV_N*C_*2;           // Wqkv^T bf16

    unsigned short* Q  = (unsigned short*)(ws + offQ);
    unsigned short* K  = (unsigned short*)(ws + offK);
    unsigned short* Vt = (unsigned short*)(ws + offV);
    unsigned short* Y  = (unsigned short*)(ws + offY);
    unsigned short* Xb = (unsigned short*)(ws + offXb);
    unsigned short* Wq = (unsigned short*)(ws + offWq);
    unsigned short* Wp = (unsigned short*)(ws + offWp);

    k_conv_x<<<((long long)M_*C_/4 + 255)/256, 256, 0, stream>>>(x, Xb);
    k_conv_w<<<(C_*QKV_N + 255)/256, 256, 0, stream>>>(W_qkv,  Wq, C_, QKV_N);
    k_conv_w<<<(C_*C_    + 255)/256, 256, 0, stream>>>(W_proj, Wp, C_, C_);
    k_vpad  <<<(B_*H_*D_*(VT_PITCH-N_) + 255)/256, 256, 0, stream>>>(Vt);

    dim3 g1((M_ + 127)/128, QKV_N/128);
    k_qkv_gemm<<<g1, 256, 0, stream>>>(Xb, Wq, b_qkv, Q, K, Vt);

    k_prompt<<<(B_*C_ + 255)/256, 256, 0, stream>>>(Vt, Y);

    k_attn<<<24192/4, 128, 0, stream>>>(Q, K, Vt, Y);

    dim3 g2((M_ + 127)/128, C_/128);
    k_proj_gemm<<<g2, 256, 0, stream>>>(Y, Wp, b_proj, (float*)d_out);
}
